// CBOW_46694884442659
// MI455X (gfx1250) — matrix operand order repaired and checked by reading, not compiled, not run on hardware
//
#include <hip/hip_runtime.h>
#include <hip/hip_bf16.h>

// ---------------------------------------------------------------------------
// CBOW with full-vocab sigmoid denominator, fused for MI455X (gfx1250).
//   h = mean(emb_v[x], axis=ctx)                     [B,E]
//   pos_b = softplus(-dot(emb_u[y_b], h_b))
//   neg_b = log( sum_v 1/(1+exp(dot(h_b, u_v))) )
//   out   = mean(pos + neg)
//
// Hot loop: 2048x50000x100 GEMM fused with sigmoid + row reduction.
//  - v_wmma_f32_16x16x32_f16 (fp16 in, fp32 acc), K padded 100->128
//  - B tiles staged block-wide in LDS via global_load_async_to_lds_b128
//    (ASYNCcnt-tracked), 4-deep rotating buffers, one barrier/iteration.
//  - All 8 ds_load_b128 pinned before the WMMA chain (asm register tie) so
//    LDS latency is paid once per tile and the 4 WMMAs issue back-to-back.
// ---------------------------------------------------------------------------

#define VOCAB 50000
#define EMB   100
#define EPAD  128
#define BATCH 2048
#define CTX   10

#define VTILES (VOCAB / 16)   // 3125, exact
#define MTILES (BATCH / 16)   // 128, exact
#define VCHUNKS 25            // gridDim.y for the GEMM kernel
#define VT_PER_CHUNK (VTILES / VCHUNKS) // 125, exact

// LDS tile: 16 rows x 128 f16, padded to 136 f16 per row (272B) so the
// per-lane 16B beats rotate over all 64 LDS banks (row stride 272 % 256 = 16).
#define LDS_ROW 136
#define NBUF 4                // rotating buffers (min for 1 barrier/iter)
                              // + 1 garbage buffer for uniform tail issues

typedef __attribute__((ext_vector_type(16))) _Float16 v16h;
typedef __attribute__((ext_vector_type(8)))  _Float16 v8h;
typedef __attribute__((ext_vector_type(8)))  float    v8f;

// ---------------------------------------------------------------------- zero
__global__ void cbow_zero_kernel(float* __restrict__ rowsum, float* __restrict__ out) {
    int i = blockIdx.x * blockDim.x + threadIdx.x;
    if (i < BATCH) rowsum[i] = 0.0f;
    if (i == 0) out[0] = 0.0f;
}

// ------------------------------------------------- h = mean(emb_v[x]) gather
__global__ __launch_bounds__(EPAD) void cbow_h_kernel(
        const int* __restrict__ x, const float* __restrict__ emb_v,
        float* __restrict__ h32, _Float16* __restrict__ h16) {
    const int b = blockIdx.x;
    const int e = threadIdx.x;          // 0..127
    float s = 0.0f;
    if (e < EMB) {
        #pragma unroll
        for (int c = 0; c < CTX; ++c) {
            int idx = x[b * CTX + c];
            s += emb_v[idx * EMB + e];
        }
        s *= (1.0f / (float)CTX);
        h32[b * EMB + e] = s;
    }
    h16[b * EPAD + e] = (e < EMB) ? (_Float16)s : (_Float16)0.0f;
}

// ------------------------------------------- emb_u f32 -> f16 padded to 128
__global__ __launch_bounds__(256) void cbow_uconv_kernel(
        const float* __restrict__ u, _Float16* __restrict__ u16) {
    int tid = blockIdx.x * 256 + threadIdx.x;
    if (tid >= VOCAB * EPAD) return;
    int v = tid >> 7;        // /128
    int e = tid & 127;
    u16[tid] = (e < EMB) ? (_Float16)u[v * EMB + e] : (_Float16)0.0f;
}

// ---------------------------------------------------------------------------
// Async copy of one 16-byte chunk of a vocab tile into LDS (ASYNCcnt++).
// dst must be an LDS address; low 32 bits of a generic LDS pointer are the
// LDS byte offset on AMDGPU.
// ---------------------------------------------------------------------------
__device__ __forceinline__ void async_b128_to_lds(const _Float16* src,
                                                  const _Float16* lds_dst) {
    unsigned           ldsa = (unsigned)(unsigned long long)(const void*)lds_dst;
    unsigned long long ga   = (unsigned long long)(const void*)src;
    asm volatile("global_load_async_to_lds_b128 %0, %1, off"
                 :: "v"(ldsa), "v"(ga) : "memory");
}

// ---------------------------------------------------------------------------
// Fused GEMM + sigmoid + row-sum.
// One wave per 16-row (M) tile; 8 waves per block share B tiles via LDS.
// D[m,n] = dot(h_m, u_n), accumulate sum_n 1/(1+exp(D[m,n])) per row.
//
// CDNA5 16-bit A-matrix 16x32 layout (per k-step of 32):
//   lanes 0-15 : row M=lane,     elems 0..7 = K {k+0..k+7},  8..15 = K {k+16..k+23}
//   lanes 16-31: row M=lane-16,  elems 0..7 = K {k+8..k+15}, 8..15 = K {k+24..k+31}
// B-matrix 32x16 layout (same K order as A):
//   lanes 0-15 : col N=lane,     elems 0..7 = K {k+0..k+7},  8..15 = K {k+16..k+23}
//   lanes 16-31: col N=lane-16,  elems 0..7 = K {k+8..k+15}, 8..15 = K {k+24..k+31}
// C/D 16x16: VGPR r, lanes 0-15 -> (M=r, N=lane); lanes 16-31 -> (M=r+8, N=lane-16)
// ---------------------------------------------------------------------------
__global__ __launch_bounds__(256) void cbow_neg_wmma_kernel(
        const _Float16* __restrict__ h16, const _Float16* __restrict__ u16,
        float* __restrict__ rowsum) {
    // 5 buffers: 4 rotating + 1 garbage sink for uniform tail prefetches.
    __shared__ __align__(32) _Float16 bbuf[NBUF + 1][16 * LDS_ROW];

    const int tid    = threadIdx.x;
    const int lane   = tid & 31;
    const int wave   = tid >> 5;
    const int m0     = (blockIdx.x * 8 + wave) * 16;   // row-tile base
    const int laneM  = lane & 15;
    const int hiHalf = (lane >> 4) & 1;                // 0: lanes 0-15, 1: 16-31

    // Per-thread async-copy slot: 256 threads x 16B = 4KB = one 16x128 f16 tile.
    const int cpRow   = tid >> 4;        // 0..15
    const int cpChunk = tid & 15;        // 0..15 (8 f16 each)
    const int cpSrcOf = cpRow * EPAD + cpChunk * 8;       // within a tile
    const int cpDstOf = cpRow * LDS_ROW + cpChunk * 8;    // padded LDS layout

    // Hoist the A fragments (invariant across the whole vocab loop): 4 k-steps.
    v16h afrag[4];
    {
        const _Float16* hrow = h16 + (m0 + laneM) * EPAD;
        #pragma unroll
        for (int ks = 0; ks < 4; ++ks) {
            const int kofs = ks * 32 + (hiHalf ? 8 : 0);
            v8h lo = *(const v8h*)(hrow + kofs);        // K kofs..kofs+7
            v8h hi = *(const v8h*)(hrow + kofs + 16);   // K kofs+16..kofs+23
            afrag[ks] = __builtin_shufflevector(lo, hi,
                0, 1, 2, 3, 4, 5, 6, 7, 8, 9, 10, 11, 12, 13, 14, 15);
        }
    }

    float acc[8];
    #pragma unroll
    for (int r = 0; r < 8; ++r) acc[r] = 0.0f;

    const int t0 = blockIdx.y * VT_PER_CHUNK;

    // Pipeline prologue: tiles 0 and 1 in flight.
    async_b128_to_lds(u16 + (t0 + 0) * 16 * EPAD + cpSrcOf, &bbuf[0][cpDstOf]);
    async_b128_to_lds(u16 + (t0 + 1) * 16 * EPAD + cpSrcOf, &bbuf[1][cpDstOf]);

    for (int t = 0; t < VT_PER_CHUNK; ++t) {
        // Issue prefetch of tile t+2 (or a harmless reload into the garbage
        // buffer near the tail, keeping exactly one issue per wave per iter
        // so the wait immediate below stays constant).
        {
            const bool live = (t + 2) < VT_PER_CHUNK;
            const int  tn   = live ? (t + 2) : t;
            const int  bn   = live ? ((t + 2) & (NBUF - 1)) : NBUF;
            async_b128_to_lds(u16 + (t0 + tn) * 16 * EPAD + cpSrcOf,
                              &bbuf[bn][cpDstOf]);
        }
        // Outstanding (in order): tile t, t+1, t+2 -> retire tile t.
        asm volatile("s_wait_asynccnt 0x2" ::: "memory");
        __syncthreads();   // publish tile t block-wide

        const _Float16* bbase = &bbuf[t & (NBUF - 1)][laneM * LDS_ROW + hiHalf * 8];

        // Materialize ALL B fragments first (one clause of 8 ds_load_b128,
        // single dscnt wait) ...
        v16h b0, b1, b2, b3;
        {
            v8h lo0 = *(const v8h*)(bbase + 0 * 32);
            v8h hi0 = *(const v8h*)(bbase + 0 * 32 + 16);
            v8h lo1 = *(const v8h*)(bbase + 1 * 32);
            v8h hi1 = *(const v8h*)(bbase + 1 * 32 + 16);
            v8h lo2 = *(const v8h*)(bbase + 2 * 32);
            v8h hi2 = *(const v8h*)(bbase + 2 * 32 + 16);
            v8h lo3 = *(const v8h*)(bbase + 3 * 32);
            v8h hi3 = *(const v8h*)(bbase + 3 * 32 + 16);
            b0 = __builtin_shufflevector(lo0, hi0, 0,1,2,3,4,5,6,7,8,9,10,11,12,13,14,15);
            b1 = __builtin_shufflevector(lo1, hi1, 0,1,2,3,4,5,6,7,8,9,10,11,12,13,14,15);
            b2 = __builtin_shufflevector(lo2, hi2, 0,1,2,3,4,5,6,7,8,9,10,11,12,13,14,15);
            b3 = __builtin_shufflevector(lo3, hi3, 0,1,2,3,4,5,6,7,8,9,10,11,12,13,14,15);
        }
        // ... and pin them live here so the scheduler cannot sink the loads
        // back between the WMMAs.
        asm volatile("" : "+v"(b0), "+v"(b1), "+v"(b2), "+v"(b3));

        v8f c = {};
        c = __builtin_amdgcn_wmma_f32_16x16x32_f16(false, afrag[0], false, b0,
                                                   (short)0, c, false, false);
        c = __builtin_amdgcn_wmma_f32_16x16x32_f16(false, afrag[1], false, b1,
                                                   (short)0, c, false, false);
        c = __builtin_amdgcn_wmma_f32_16x16x32_f16(false, afrag[2], false, b2,
                                                   (short)0, c, false, false);
        c = __builtin_amdgcn_wmma_f32_16x16x32_f16(false, afrag[3], false, b3,
                                                   (short)0, c, false, false);

        // sigmoid(-score) = 1/(1+exp(score)); accumulate per (row, lane-col)
        #pragma unroll
        for (int r = 0; r < 8; ++r)
            acc[r] += __builtin_amdgcn_rcpf(1.0f + __expf(c[r]));
    }

    // Reduce over the 16 N-columns held in each 16-lane half.
    #pragma unroll
    for (int r = 0; r < 8; ++r) {
        float s = acc[r];
        s += __shfl_xor(s, 1, 32);
        s += __shfl_xor(s, 2, 32);
        s += __shfl_xor(s, 4, 32);
        s += __shfl_xor(s, 8, 32);
        if (laneM == 0)
            atomicAdd(rowsum + m0 + r + (hiHalf ? 8 : 0), s);
    }
}

// ------------------------------------------------------ finalize: mean(pos+neg)
__global__ __launch_bounds__(256) void cbow_final_kernel(
        const int* __restrict__ y, const float* __restrict__ emb_u,
        const float* __restrict__ h32, const float* __restrict__ rowsum,
        float* __restrict__ out) {
    __shared__ float red[256];
    const int b = blockIdx.x * 256 + threadIdx.x;   // BATCH == 8*256

    const float* urow = emb_u + y[b] * EMB;
    const float* hrow = h32 + b * EMB;
    float s = 0.0f;
    #pragma unroll 4
    for (int e = 0; e < EMB; ++e) s += urow[e] * hrow[e];

    // pos = -log_sigmoid(s) = softplus(-s), numerically stable
    const float xs  = -s;
    const float pos = fmaxf(xs, 0.0f) + log1pf(__expf(-fabsf(xs)));
    const float neg = logf(rowsum[b]);

    red[threadIdx.x] = (pos + neg) * (1.0f / (float)BATCH);
    __syncthreads();
    for (int stride = 128; stride > 0; stride >>= 1) {
        if (threadIdx.x < stride) red[threadIdx.x] += red[threadIdx.x + stride];
        __syncthreads();
    }
    if (threadIdx.x == 0) atomicAdd(out, red[0]);
}

// ---------------------------------------------------------------------------
extern "C" void kernel_launch(void* const* d_in, const int* in_sizes, int n_in,
                              void* d_out, int out_size, void* d_ws, size_t ws_size,
                              hipStream_t stream) {
    const int*   x     = (const int*)  d_in[0];   // [BATCH, CTX]
    const int*   y     = (const int*)  d_in[1];   // [BATCH]
    const float* emb_v = (const float*)d_in[2];   // [VOCAB, EMB]
    const float* emb_u = (const float*)d_in[3];   // [VOCAB, EMB]
    float*       out   = (float*)d_out;

    // Workspace carve-up (all offsets 256B aligned):
    char* ws = (char*)d_ws;
    float*    rowsum = (float*)   (ws);                          //  2048 f32
    float*    h32    = (float*)   (ws + 8192);                   //  2048*100 f32
    _Float16* h16    = (_Float16*)(ws + 8192 + 819200);          //  2048*128 f16
    _Float16* u16    = (_Float16*)(ws + 8192 + 819200 + 524288); // 50000*128 f16
    // total ~14.2 MB

    // 1) zero accumulators (required every call; we accumulate via atomics)
    cbow_zero_kernel<<<(BATCH + 255) / 256, 256, 0, stream>>>(rowsum, out);

    // 2) h = mean(emb_v[x]) in f32 + padded f16
    cbow_h_kernel<<<BATCH, EPAD, 0, stream>>>(x, emb_v, h32, h16);

    // 3) emb_u -> padded f16
    cbow_uconv_kernel<<<(VOCAB * EPAD + 255) / 256, 256, 0, stream>>>(emb_u, u16);

    // 4) fused WMMA GEMM + sigmoid + row reduction (async-LDS pipelined)
    dim3 grid(MTILES / 8, VCHUNKS);   // 16 x 25 = 400 workgroups, 8 waves each
    cbow_neg_wmma_kernel<<<grid, 256, 0, stream>>>(h16, u16, rowsum);

    // 5) finalize scalar
    cbow_final_kernel<<<BATCH / 256, 256, 0, stream>>>(y, emb_u, h32, rowsum, out);
}
